// GlobalAttention_44358422233601
// MI455X (gfx1250) — matrix operand order repaired and checked by reading, not compiled, not run on hardware
//
#include <hip/hip_runtime.h>

// Problem dims (fixed by reference)
constexpr int NB = 32;    // batch
constexpr int NT = 1024;  // T
constexpr int NS = 1024;  // S
constexpr int ND = 1024;  // D

typedef __attribute__((ext_vector_type(16))) __bf16    v16bf;
typedef __attribute__((ext_vector_type(8)))  __bf16    v8bf;
typedef __attribute__((ext_vector_type(8)))  float     v8f;
typedef __attribute__((ext_vector_type(4)))  unsigned  v4u;
typedef __attribute__((ext_vector_type(8)))  int       v8i;
typedef __attribute__((ext_vector_type(4)))  int       v4i;

// Explicit LDS (addrspace 3) types so shared-memory accesses lower to ds_* ops.
typedef __attribute__((address_space(3))) __bf16 lds_bf;
typedef __attribute__((address_space(3))) v8bf   lds_v8bf;
typedef __attribute__((address_space(3))) v4u    lds_v4u;

#define LDA 40                       // LDS row stride in bf16 (32 data + 8 pad => 80B rows)
#define TILE_ELEMS (128 * LDA)       // one 128x32 bf16 tile in LDS (incl. padding)
#define TILE_BYTES (TILE_ELEMS * 2)

#if defined(__HIP_DEVICE_COMPILE__) && __has_builtin(__builtin_amdgcn_tensor_load_to_lds)
#define USE_TDM 1
#else
#define USE_TDM 0
#endif

// ---- helpers -------------------------------------------------------------

__device__ __forceinline__ unsigned pk2(float x, float y) {
  unsigned short a = __builtin_bit_cast(unsigned short, (__bf16)x);
  unsigned short b = __builtin_bit_cast(unsigned short, (__bf16)y);
  return (unsigned)a | ((unsigned)b << 16);
}

#if USE_TDM
// LDS byte offset of an addrspace(3) pointer.
__device__ __forceinline__ unsigned lds_off3(const lds_bf* p) {
  return (unsigned)(size_t)p;
}

// TDM: async-load a 128(row) x 32(bf16) tile, row stride `strideElems`, into LDS
// with 16B padding appended after every 64B row (reproduces the LDA=40 layout).
__device__ __forceinline__ void tdm_load_tile(unsigned lds_addr, const __bf16* g,
                                              unsigned strideElems) {
  unsigned long long ga = (unsigned long long)(size_t)g;
  v4u g0;
  g0[0] = 1u;                                                   // count=1 (valid descriptor)
  g0[1] = lds_addr;                                             // LDS byte address
  g0[2] = (unsigned)ga;                                         // global_addr[31:0]
  g0[3] = (unsigned)((ga >> 32) & 0x01FFFFFFull) | 0x80000000u; // addr[56:32] | type=2
  v8i g1;
  g1[0] = (int)0x06D10000u;   // data_size=2B | pad_enable | pad_interval=16dw | pad_amount=4dw
  g1[1] = (int)(32u << 16);   // tensor_dim0 = 32 (low 16 in [31:16])
  g1[2] = (int)(128u << 16);  // tensor_dim0 hi=0 | tensor_dim1 = 128 (low 16)
  g1[3] = (int)(32u << 16);   // tensor_dim1 hi=0 | tile_dim0 = 32
  g1[4] = (int)128;           // tile_dim1 = 128 | tile_dim2 = 0
  g1[5] = (int)strideElems;   // tensor_dim0_stride[31:0]
  g1[6] = 0;                  // stride0 hi | stride1 lo
  g1[7] = 0;                  // stride1 hi
  v4i z4 = {0, 0, 0, 0};
#if defined(__clang_major__) && __clang_major__ >= 23
  v8i z8 = {0, 0, 0, 0, 0, 0, 0, 0};
  __builtin_amdgcn_tensor_load_to_lds(g0, g1, z4, z4, z8, 0);
#else
  __builtin_amdgcn_tensor_load_to_lds(g0, g1, z4, z4, 0);
#endif
}
#endif  // USE_TDM

// Fallback staging: copy a 128x32 bf16 tile (row stride strideElems) into LDS.
__device__ __forceinline__ void stage_direct(lds_bf* lds, const __bf16* g,
                                             unsigned strideElems, int tid) {
  const int r  = tid >> 1;
  const int kh = (tid & 1) << 4;
  const uint4* s = (const uint4*)(g + (size_t)r * strideElems + kh);
  uint4 a = s[0], b = s[1];
  v4u q0 = { a.x, a.y, a.z, a.w };
  v4u q1 = { b.x, b.y, b.z, b.w };
  lds_v4u* d = (lds_v4u*)(lds + r * LDA + kh);
  d[0] = q0;
  d[1] = q1;
}

// A fragment (16x32 bf16): lanes 0-15 hold M=lane, K=0..7/16..23; lanes 16-31 K=8..15/24..31.
__device__ __forceinline__ v16bf load_a_frag(const lds_bf* lA, int mBase, int lane) {
  const int m  = mBase + (lane & 15);
  const int ks = (lane >> 4) << 3;
  const lds_bf* p = lA + m * LDA + ks;
  v8bf lo = *(const lds_v8bf*)(p);
  v8bf hi = *(const lds_v8bf*)(p + 16);
  v16bf f;
#pragma unroll
  for (int i = 0; i < 8; ++i) { f[i] = lo[i]; f[i + 8] = hi[i]; }
  return f;
}

// B fragment (32x16 bf16): lanes 0-15 hold N=lane, K=0..7/16..23; lanes 16-31 K=8..15/24..31.
__device__ __forceinline__ v16bf load_b_frag(const lds_bf* lB, int nBase, int lane) {
  const int n  = nBase + (lane & 15);
  const int ks = (lane >> 4) << 3;
  const lds_bf* p = lB + n * LDA + ks;
  v8bf lo = *(const lds_v8bf*)(p);
  v8bf hi = *(const lds_v8bf*)(p + 16);
  v16bf f;
#pragma unroll
  for (int i = 0; i < 8; ++i) { f[i] = lo[i]; f[i + 8] = hi[i]; }
  return f;
}

__device__ __forceinline__ v8f wmma_bf16(v16bf a, v16bf b, v8f c) {
  return __builtin_amdgcn_wmma_f32_16x16x32_bf16(false, a, false, b, (short)0, c, false, false);
}

// Shared 128x128 GEMM mainloop: TDM double-buffered staging + 8 WMMAs/wave/K-step.
__device__ __forceinline__ void gemm_core(const __bf16* __restrict__ gA, unsigned strideA,
                                          const __bf16* __restrict__ gB, unsigned strideB,
                                          int K, __bf16* ldsGeneric, int tid, v8f acc[4][2]) {
  const int lane = tid & 31;
  const int wave = tid >> 5;
  const int mW = (wave >> 2) << 6;  // 0 / 64
  const int nW = (wave & 3) << 5;   // 0 / 32 / 64 / 96

  lds_bf* lbase = (lds_bf*)ldsGeneric;   // generic -> addrspace(3)

#if USE_TDM
  const unsigned o0  = lds_off3(lbase);
  const unsigned oA0 = o0,                  oA1 = o0 + TILE_BYTES;
  const unsigned oB0 = o0 + 2 * TILE_BYTES, oB1 = o0 + 3 * TILE_BYTES;
  if (wave == 0) {                       // one wave drives the Tensor Data Mover
    tdm_load_tile(oA0, gA, strideA);
    tdm_load_tile(oB0, gB, strideB);
  }
#endif

#pragma unroll 1
  for (int kb = 0; kb < K; kb += 32) {
    const int cur = (kb >> 5) & 1;
    lds_bf* lA = lbase + (cur ? TILE_ELEMS : 0);
    lds_bf* lB = lbase + 2 * TILE_ELEMS + (cur ? TILE_ELEMS : 0);
    __syncthreads();                     // everyone done reading buf[cur^1]
#if USE_TDM
    if (wave == 0) {
      if (kb + 32 < K) {                 // prefetch next pair into the other buffer
        tdm_load_tile(cur ? oA0 : oA1, gA + kb + 32, strideA);
        tdm_load_tile(cur ? oB0 : oB1, gB + kb + 32, strideB);
        __builtin_amdgcn_s_wait_tensorcnt(2);  // current pair landed; next pair in flight
      } else {
        __builtin_amdgcn_s_wait_tensorcnt(0);
      }
    }
#else
    stage_direct(lA, gA + kb, strideA, tid);
    stage_direct(lB, gB + kb, strideB, tid);
#endif
    __syncthreads();                     // tile visible to all waves

    v16bf af[4];
#pragma unroll
    for (int i = 0; i < 4; ++i) af[i] = load_a_frag(lA, mW + (i << 4), lane);
    v16bf bfr[2];
#pragma unroll
    for (int j = 0; j < 2; ++j) bfr[j] = load_b_frag(lB, nW + (j << 4), lane);
#pragma unroll
    for (int i = 0; i < 4; ++i)
#pragma unroll
      for (int j = 0; j < 2; ++j)
        acc[i][j] = wmma_bf16(af[i], bfr[j], acc[i][j]);
  }
}

// ---- kernel 0a: f32 -> bf16 bulk convert (n8 = elements/8) ----

__global__ __launch_bounds__(256) void k_cvt(const float* __restrict__ src,
                                             __bf16* __restrict__ dst, int n8) {
  const int i = blockIdx.x * 256 + threadIdx.x;
  if (i >= n8) return;
  const float4* s4 = (const float4*)src;
  float4 a = s4[2 * i], b = s4[2 * i + 1];
  uint4 q = { pk2(a.x, a.y), pk2(a.z, a.w), pk2(b.x, b.y), pk2(b.z, b.w) };
  ((uint4*)dst)[i] = q;
}

// ---- kernel 0b: context (b,s,d) f32 -> ctxT (b,d,s) bf16, tiled transpose ----

__global__ __launch_bounds__(256) void k_transpose(const float* __restrict__ src,
                                                   __bf16* __restrict__ dst) {
  __shared__ float tile[32][33];
  const int b  = blockIdx.z;
  const int s0 = blockIdx.x << 5;
  const int d0 = blockIdx.y << 5;
  const int tid = threadIdx.x;
  const int r  = tid >> 3;          // 0..31
  const int c4 = (tid & 7) << 2;    // 0..28
  const float4 v = *(const float4*)(src + ((size_t)b * NS + s0 + r) * ND + d0 + c4);
  tile[r][c4 + 0] = v.x; tile[r][c4 + 1] = v.y;
  tile[r][c4 + 2] = v.z; tile[r][c4 + 3] = v.w;
  __syncthreads();
  uint2 q;
  q.x = pk2(tile[c4 + 0][r], tile[c4 + 1][r]);
  q.y = pk2(tile[c4 + 2][r], tile[c4 + 3][r]);
  *(uint2*)(dst + ((size_t)b * ND + d0 + r) * NS + s0 + c4) = q;
}

// ---- kernel 1: raw scores align[t,b,s] = sum_d input[b,t,d]*context[b,s,d] ----

__global__ __launch_bounds__(256) void ga_scores(const __bf16* __restrict__ qbf,
                                                 const __bf16* __restrict__ kbf,
                                                 float* __restrict__ alignOut) {
  __shared__ __bf16 lds[4 * TILE_ELEMS];
  const int b  = blockIdx.z;
  const int t0 = blockIdx.x << 7;
  const int s0 = blockIdx.y << 7;
  const int tid  = threadIdx.x;
  const int lane = tid & 31;
  const int wave = tid >> 5;
  const int mW = (wave >> 2) << 6;
  const int nW = (wave & 3) << 5;

  v8f acc[4][2] = {};
  gemm_core(qbf + ((size_t)b * NT + t0) * ND, ND,
            kbf + ((size_t)b * NS + s0) * ND, ND, ND, lds, tid, acc);

  const int col  = lane & 15;
  const int rsel = (lane >> 4) << 3;
#pragma unroll
  for (int i = 0; i < 4; ++i)
#pragma unroll
    for (int j = 0; j < 2; ++j) {
      const int s = s0 + nW + (j << 4) + col;
#pragma unroll
      for (int e = 0; e < 8; ++e) {
        const int t = t0 + mW + (i << 4) + rsel + e;
        alignOut[((size_t)t * NB + b) * NS + s] = acc[i][j][e];
      }
    }
}

// ---- kernel 2: softmax over each (t,b) row; writes f32 (d_out) + bf16 (for next GEMM) ----

__global__ __launch_bounds__(256) void ga_softmax(float* __restrict__ p,
                                                  __bf16* __restrict__ pbf) {
  __shared__ float red[256];
  float* row = p + (size_t)blockIdx.x * NS;
  const int tid = threadIdx.x;
  float4 x = ((const float4*)row)[tid];
  float m = fmaxf(fmaxf(x.x, x.y), fmaxf(x.z, x.w));
  red[tid] = m;
  __syncthreads();
  for (int off = 128; off > 0; off >>= 1) {
    if (tid < off) red[tid] = fmaxf(red[tid], red[tid + off]);
    __syncthreads();
  }
  m = red[0];
  __syncthreads();
  x.x = __expf(x.x - m); x.y = __expf(x.y - m);
  x.z = __expf(x.z - m); x.w = __expf(x.w - m);
  red[tid] = x.x + x.y + x.z + x.w;
  __syncthreads();
  for (int off = 128; off > 0; off >>= 1) {
    if (tid < off) red[tid] += red[tid + off];
    __syncthreads();
  }
  const float inv = 1.0f / red[0];
  x.x *= inv; x.y *= inv; x.z *= inv; x.w *= inv;
  ((float4*)row)[tid] = x;
  uint2 q = { pk2(x.x, x.y), pk2(x.z, x.w) };
  *(uint2*)(pbf + (size_t)blockIdx.x * NS + (tid << 2)) = q;
}

// ---- kernel 3: c[b,t,d] = sum_s P[t,b,s]*context[b,s,d]  (B operand = pre-transposed ctxT) ----

__global__ __launch_bounds__(256) void ga_context(const __bf16* __restrict__ pbf,
                                                  const __bf16* __restrict__ ktbf,
                                                  __bf16* __restrict__ cbf) {
  __shared__ __bf16 lds[4 * TILE_ELEMS];
  const int b  = blockIdx.z;
  const int t0 = blockIdx.x << 7;
  const int d0 = blockIdx.y << 7;
  const int tid  = threadIdx.x;
  const int lane = tid & 31;
  const int wave = tid >> 5;
  const int mW = (wave >> 2) << 6;
  const int nW = (wave & 3) << 5;

  v8f acc[4][2] = {};
  gemm_core(pbf + ((size_t)t0 * NB + b) * NS, NB * NS,
            ktbf + ((size_t)b * ND + d0) * NS, NS, NS, lds, tid, acc);

  const int col  = lane & 15;
  const int rsel = (lane >> 4) << 3;
#pragma unroll
  for (int i = 0; i < 4; ++i)
#pragma unroll
    for (int j = 0; j < 2; ++j) {
      const int d = d0 + nW + (j << 4) + col;
#pragma unroll
      for (int e = 0; e < 8; ++e) {
        const int t = t0 + mW + (i << 4) + rsel + e;
        cbf[((size_t)b * NT + t) * ND + d] = (__bf16)acc[i][j][e];
      }
    }
}

// ---- kernel 4: attn[t,b,e] = tanh( c.W_c^T + x.W_i^T ) as one K=2*ND GEMM in two passes ----

__global__ __launch_bounds__(256) void ga_output(const __bf16* __restrict__ cbf,
                                                 const __bf16* __restrict__ qbf,
                                                 const __bf16* __restrict__ wbf,
                                                 float* __restrict__ attnOut) {
  __shared__ __bf16 lds[4 * TILE_ELEMS];
  const int b  = blockIdx.z;
  const int t0 = blockIdx.x << 7;
  const int e0 = blockIdx.y << 7;
  const int tid  = threadIdx.x;
  const int lane = tid & 31;
  const int wave = tid >> 5;
  const int mW = (wave >> 2) << 6;
  const int nW = (wave & 3) << 5;

  v8f acc[4][2] = {};
  // k in [0, ND): A = c
  gemm_core(cbf + ((size_t)b * NT + t0) * ND, ND,
            wbf + (size_t)e0 * (2 * ND), 2 * ND, ND, lds, tid, acc);
  // k in [ND, 2*ND): A = input
  gemm_core(qbf + ((size_t)b * NT + t0) * ND, ND,
            wbf + (size_t)e0 * (2 * ND) + ND, 2 * ND, ND, lds, tid, acc);

  const int col  = lane & 15;
  const int rsel = (lane >> 4) << 3;
#pragma unroll
  for (int i = 0; i < 4; ++i)
#pragma unroll
    for (int j = 0; j < 2; ++j) {
      const int e = e0 + nW + (j << 4) + col;
#pragma unroll
      for (int ee = 0; ee < 8; ++ee) {
        const int t = t0 + mW + (i << 4) + rsel + ee;
        attnOut[((size_t)t * NB + b) * ND + e] = tanhf(acc[i][j][ee]);
      }
    }
}

// ---- host glue -----------------------------------------------------------

extern "C" void kernel_launch(void* const* d_in, const int* in_sizes, int n_in,
                              void* d_out, int out_size, void* d_ws, size_t ws_size,
                              hipStream_t stream) {
  (void)in_sizes; (void)n_in; (void)out_size; (void)ws_size;
  const float* inp  = (const float*)d_in[0];   // (B,T,D) f32
  const float* ctx  = (const float*)d_in[1];   // (B,S,D) f32
  const float* wout = (const float*)d_in[2];   // (D,2D)  f32

  float* attn   = (float*)d_out;                     // (T,B,D)
  float* alignv = attn + (size_t)NT * NB * ND;       // (T,B,S)

  const size_t E1 = (size_t)NB * NT * ND;            // 33.5M elems
  __bf16* qbf  = (__bf16*)d_ws;                      // input  bf16 (b,t,d)
  __bf16* kbf  = qbf  + E1;                          // context bf16 (b,s,d)
  __bf16* ktbf = kbf  + E1;                          // context bf16 transposed (b,d,s)
  __bf16* pbf  = ktbf + E1;                          // softmax bf16 (t,b,s)
  __bf16* cbf  = pbf  + E1;                          // c bf16 (b,t,d)
  __bf16* wbf  = cbf  + E1;                          // W_out bf16 (2*ND per row)

  dim3 blk(256);
  k_cvt<<<dim3((int)(E1 / 8 / 256)), blk, 0, stream>>>(inp, qbf, (int)(E1 / 8));
  k_cvt<<<dim3((int)(E1 / 8 / 256)), blk, 0, stream>>>(ctx, kbf, (int)(E1 / 8));
  k_cvt<<<dim3((int)((size_t)ND * 2 * ND / 8 / 256)), blk, 0, stream>>>(
      wout, wbf, (int)((size_t)ND * 2 * ND / 8));
  k_transpose<<<dim3(NS / 32, ND / 32, NB), blk, 0, stream>>>(ctx, ktbf);

  ga_scores <<<dim3(NT / 128, NS / 128, NB), blk, 0, stream>>>(qbf, kbf, alignv);
  ga_softmax<<<dim3(NT * NB), blk, 0, stream>>>(alignv, pbf);
  ga_context<<<dim3(NT / 128, ND / 128, NB), blk, 0, stream>>>(pbf, ktbf, cbf);
  ga_output <<<dim3(NT / 128, ND / 128, NB), blk, 0, stream>>>(cbf, qbf, wbf, attn);
}
